// MPNNModel_3470333575711
// MI455X (gfx1250) — matrix operand order repaired and checked by reading, not compiled, not run on hardware
//
#include <hip/hip_runtime.h>
#include <hip/hip_bf16.h>
#include <stdint.h>

// MPNN on gfx1250: GEMMs via V_WMMA_F32_16X16X32_BF16 (f32 accum),
// contiguous A-tiles staged to LDS via the Tensor Data Mover (TDM),
// weight B-fragments batch-loaded per k-slice so WMMAs issue back-to-back.

typedef __attribute__((ext_vector_type(16))) __bf16   v16bf;
typedef __attribute__((ext_vector_type(8)))  __bf16   v8bf;
typedef __attribute__((ext_vector_type(8)))  float    v8f;
typedef __attribute__((ext_vector_type(4)))  uint32_t v4u;
typedef __attribute__((ext_vector_type(8)))  int      v8i;
typedef __attribute__((ext_vector_type(4)))  int      v4i;

#define N_NODES  20000
#define N_EDGES  320000
#define EMB      128
#define IN_DIM   11
#define N_LAYERS 4
#define NUM_GRAPHS 64

__device__ __forceinline__ float lrelu(float v) { return v > 0.0f ? v : 0.01f * v; }

// ---- TDM: 1-D contiguous tile -> LDS.  D# layout per CDNA5 ISA.
// dataSizeCode: 0=1B,1=2B,2=4B.  tileElems <= 65535.  tensorElems bounds OOB (reads
// beyond return zero).  One DMA per wave; completion via TENSORcnt.
__device__ __forceinline__ void tdm_load_1d(const void* gsrc, uint32_t lds_off,
                                            uint32_t tileElems, uint32_t tensorElems,
                                            uint32_t dataSizeCode) {
  const uint64_t ga = (uint64_t)(uintptr_t)gsrc;
  v4u g0;
  g0[0] = 1u;                                            // count=1, user mode
  g0[1] = lds_off;                                       // lds_addr (bytes)
  g0[2] = (uint32_t)ga;                                  // global_addr[31:0]
  g0[3] = (uint32_t)((ga >> 32) & 0x01FFFFFFu) | 0x80000000u; // addr[56:32] | type=2
  v8i g1;
  g1[0] = (int)(dataSizeCode << 16);                     // data_size; no multicast/pad
  g1[1] = (int)((tensorElems & 0xFFFFu) << 16);          // tensor_dim0[15:0] @ bits 63:48
  g1[2] = (int)(((tensorElems >> 16) & 0xFFFFu) | (1u << 16)); // dim0 hi | tensor_dim1=1
  g1[3] = (int)((tileElems & 0xFFFFu) << 16);            // tile_dim0 @ bits 127:112
  g1[4] = 1;                                             // tile_dim1=1
  g1[5] = (int)tensorElems;                              // tensor_dim0_stride lo32
  g1[6] = 0;
  g1[7] = 0;
  v4i z4 = {};
#if defined(__clang_major__) && (__clang_major__ >= 23)
  v8i z8 = {};
  __builtin_amdgcn_tensor_load_to_lds(g0, g1, z4, z4, z8, 0);
#else
  __builtin_amdgcn_tensor_load_to_lds(g0, g1, z4, z4, 0);
#endif
}

// Generic shared pointer -> LDS byte offset (ISA §10.2: LDS aperture low 32 bits).
__device__ __forceinline__ uint32_t lds_off32(const void* p) {
  return (uint32_t)(uintptr_t)p;
}

// A-fragment (16x32 bf16, MxK), ISA wave32 layout: lane<16 holds K={0..7,16..23},
// lane>=16 holds K={8..15,24..31}; p = row + kslice*32 + kb, kb=(lane>=16)?8:0.
__device__ __forceinline__ v16bf frag_bf16(const __bf16* p) {
  v8bf lo = *(const v8bf*)(p);
  v8bf hi = *(const v8bf*)(p + 16);
  return __builtin_shufflevector(lo, hi, 0,1,2,3,4,5,6,7,8,9,10,11,12,13,14,15);
}

__device__ __forceinline__ v16bf frag_f32(const float* p) {
  float4 a = *(const float4*)(p);
  float4 b = *(const float4*)(p + 4);
  float4 c = *(const float4*)(p + 16);
  float4 d = *(const float4*)(p + 20);
  v16bf r;
  r[0]=(__bf16)a.x;  r[1]=(__bf16)a.y;  r[2]=(__bf16)a.z;  r[3]=(__bf16)a.w;
  r[4]=(__bf16)b.x;  r[5]=(__bf16)b.y;  r[6]=(__bf16)b.z;  r[7]=(__bf16)b.w;
  r[8]=(__bf16)c.x;  r[9]=(__bf16)c.y;  r[10]=(__bf16)c.z; r[11]=(__bf16)c.w;
  r[12]=(__bf16)d.x; r[13]=(__bf16)d.y; r[14]=(__bf16)d.z; r[15]=(__bf16)d.w;
  return r;
}

// Pack weight W[L][O][I] (f32) into B fragments:
// P[((tn*KS+tk)*32+lane)*16 + j] = bf16(W[l][tn*16+(lane&15)][tk*32+8*(lane>>4)+j+(j&8)])
__global__ void pack_kernel(const float* __restrict__ W, __bf16* __restrict__ P,
                            int L, int O, int I) {
  const int KS = I >> 5;
  const size_t per = (size_t)O * I;
  const size_t total = (size_t)L * per;
  for (size_t idx = (size_t)blockIdx.x * blockDim.x + threadIdx.x; idx < total;
       idx += (size_t)gridDim.x * blockDim.x) {
    const size_t l = idx / per;
    size_t rem = idx % per;
    const int j = (int)(rem & 15);  rem >>= 4;
    const int lane = (int)(rem & 31); rem >>= 5;
    const int tk = (int)(rem % KS);
    const int tn = (int)(rem / KS);
    P[idx] = (__bf16)W[l * per + (size_t)(tn * 16 + (lane & 15)) * I + tk * 32 + ((lane >> 4) << 3) + j + (j & 8)];
  }
}

__global__ void zero_kernel(float* __restrict__ p, long n) {
  long i = (long)blockIdx.x * blockDim.x + threadIdx.x;
  const long stride = (long)gridDim.x * blockDim.x;
  for (; i < n; i += stride) p[i] = 0.0f;
}

__global__ void lin_in_kernel(const float* __restrict__ x, const float* __restrict__ W,
                              const float* __restrict__ b,
                              float* __restrict__ h, __bf16* __restrict__ hbf) {
  const int node = blockIdx.x;
  const int o = threadIdx.x;
  if (node >= N_NODES) return;
  float s = b[o];
  const float* xr = x + (size_t)node * IN_DIM;
  const float* wr = W + (size_t)o * IN_DIM;
  #pragma unroll
  for (int k = 0; k < IN_DIM; ++k) s += xr[k] * wr[k];
  const size_t off = (size_t)node * EMB + o;
  h[off] = s;
  hbf[off] = (__bf16)s;
}

// Fused edge MLP + scatter-add.  One wave per 16 edges, 8 waves/block.
__global__ void __launch_bounds__(256)
edge_msg_kernel(const __bf16* __restrict__ hbf, const float* __restrict__ eattr,
                const int* __restrict__ srcI, const int* __restrict__ dstI,
                const __bf16* __restrict__ M1p, const float* __restrict__ M1b,
                const __bf16* __restrict__ M2p, const float* __restrict__ M2b,
                float* __restrict__ aggr) {
  __shared__ __align__(16) __bf16 smsg[8][16][EMB];   // 32 KB activation tiles
  __shared__ __align__(16) float  seat[8][16][EMB];   // 64 KB edge_attr tiles (TDM dest)
  __shared__ int sdst[8][16];
  const int wave = threadIdx.x >> 5;
  const int lane = threadIdx.x & 31;
  const int e0 = (blockIdx.x * 8 + wave) * 16;
  if (e0 >= N_EDGES) return;                    // wave-uniform
  const int r  = lane & 15;
  const int kb = (lane >> 4) << 3;

  // Kick off async TDM of this wave's contiguous [16 x 128] f32 edge_attr tile.
  tdm_load_1d(eattr + (size_t)e0 * EMB, lds_off32(&seat[wave][0][0]),
              16 * EMB, 16 * EMB, /*4B*/2);

  int e = e0 + r; if (e >= N_EDGES) e = N_EDGES - 1;
  const int di = dstI[e];
  const int si = srcI[e];
  if (lane < 16) sdst[wave][r] = di;
  const __bf16* rowi = hbf + (size_t)di * EMB;   // x_i = h[dst]
  const __bf16* rowj = hbf + (size_t)si * EMB;   // x_j = h[src]

  const v8f zero = {};
  v8f acc[8];
  #pragma unroll
  for (int t = 0; t < 8; ++t) acc[t] = zero;

  // GEMM1 k-slices 0..7: x_i / x_j gathers (overlap with TDM in flight)
  #pragma unroll
  for (int tk = 0; tk < 8; ++tk) {
    const int kc = ((tk & 3) << 5) + kb;
    v16bf a;
    if (tk < 4) a = frag_bf16(rowi + kc);
    else        a = frag_bf16(rowj + kc);
    v16bf b[8];
    #pragma unroll
    for (int t = 0; t < 8; ++t)
      b[t] = *(const v16bf*)(M1p + (((size_t)(t * 12 + tk)) * 32 + lane) * 16);
    #pragma unroll
    for (int t = 0; t < 8; ++t)
      acc[t] = __builtin_amdgcn_wmma_f32_16x16x32_bf16(false, a, false, b[t],
                                                       (short)0, acc[t], false, false);
  }

  // GEMM1 k-slices 8..11: edge_attr from LDS (wait for TDM)
  __builtin_amdgcn_s_wait_tensorcnt((short)0);
  __builtin_amdgcn_wave_barrier();
  #pragma unroll
  for (int tk2 = 0; tk2 < 4; ++tk2) {
    const int tk = 8 + tk2;
    v16bf a = frag_f32(&seat[wave][r][(tk2 << 5) + kb]);
    v16bf b[8];
    #pragma unroll
    for (int t = 0; t < 8; ++t)
      b[t] = *(const v16bf*)(M1p + (((size_t)(t * 12 + tk)) * 32 + lane) * 16);
    #pragma unroll
    for (int t = 0; t < 8; ++t)
      acc[t] = __builtin_amdgcn_wmma_f32_16x16x32_bf16(false, a, false, b[t],
                                                       (short)0, acc[t], false, false);
  }

  // bias + leaky, stage bf16 tile (C/D layout: n = t*16+r, m = v+kb)
  #pragma unroll
  for (int t = 0; t < 8; ++t) {
    const int n = t * 16 + r;
    const float bs = M1b[n];
    #pragma unroll
    for (int v = 0; v < 8; ++v)
      smsg[wave][v + kb][n] = (__bf16)lrelu(acc[t][v] + bs);
  }
  __builtin_amdgcn_wave_barrier();

  // GEMM2: [16x128] @ [128x128]
  v8f acc2[8];
  #pragma unroll
  for (int t = 0; t < 8; ++t) acc2[t] = zero;
  #pragma unroll
  for (int tk = 0; tk < 4; ++tk) {
    v16bf a = frag_bf16(&smsg[wave][r][(tk << 5) + kb]);
    v16bf b[8];
    #pragma unroll
    for (int t = 0; t < 8; ++t)
      b[t] = *(const v16bf*)(M2p + (((size_t)(t * 4 + tk)) * 32 + lane) * 16);
    #pragma unroll
    for (int t = 0; t < 8; ++t)
      acc2[t] = __builtin_amdgcn_wmma_f32_16x16x32_bf16(false, a, false, b[t],
                                                        (short)0, acc2[t], false, false);
  }

  // bias + leaky + scatter-add into aggr[dst]
  #pragma unroll
  for (int t = 0; t < 8; ++t) {
    const int n = t * 16 + r;
    const float bs = M2b[n];
    #pragma unroll
    for (int v = 0; v < 8; ++v) {
      const int m = v + kb;
      if (e0 + m < N_EDGES)
        atomicAdd(&aggr[(size_t)sdst[wave][m] * EMB + n], lrelu(acc2[t][v] + bs));
    }
  }
}

// h += leaky([h | aggr] @ U.T + b).  One wave per 16 nodes; A tiles via TDM.
__global__ void __launch_bounds__(256)
node_update_kernel(float* __restrict__ h, __bf16* __restrict__ hbf,
                   const float* __restrict__ aggr,
                   const __bf16* __restrict__ Up, const float* __restrict__ Ub) {
  __shared__ __align__(16) float  sagg[8][16][EMB];   // 64 KB
  __shared__ __align__(16) __bf16 shh[8][16][EMB];    // 32 KB
  const int wave = threadIdx.x >> 5;
  const int lane = threadIdx.x & 31;
  const int n0 = (blockIdx.x * 8 + wave) * 16;
  if (n0 >= N_NODES) return;
  const int r  = lane & 15;
  const int kb = (lane >> 4) << 3;

  // Contiguous [16 x 128] tiles of h(bf16) and aggr(f32); tensor_dim bounds the
  // ragged last tile (TDM returns zeros OOB, results masked on store).
  const uint32_t validRows = (uint32_t)((N_NODES - n0) < 16 ? (N_NODES - n0) : 16);
  tdm_load_1d(hbf  + (size_t)n0 * EMB, lds_off32(&shh[wave][0][0]),
              16 * EMB, validRows * EMB, /*2B*/1);
  tdm_load_1d(aggr + (size_t)n0 * EMB, lds_off32(&sagg[wave][0][0]),
              16 * EMB, validRows * EMB, /*4B*/2);

  const v8f zero = {};
  v8f acc[8];
  #pragma unroll
  for (int t = 0; t < 8; ++t) acc[t] = zero;

  __builtin_amdgcn_s_wait_tensorcnt((short)0);
  __builtin_amdgcn_wave_barrier();

  #pragma unroll
  for (int tk = 0; tk < 8; ++tk) {
    v16bf a;
    if (tk < 4) a = frag_bf16(&shh[wave][r][(tk << 5) + kb]);
    else        a = frag_f32(&sagg[wave][r][((tk - 4) << 5) + kb]);
    v16bf b[8];
    #pragma unroll
    for (int t = 0; t < 8; ++t)
      b[t] = *(const v16bf*)(Up + (((size_t)(t * 8 + tk)) * 32 + lane) * 16);
    #pragma unroll
    for (int t = 0; t < 8; ++t)
      acc[t] = __builtin_amdgcn_wmma_f32_16x16x32_bf16(false, a, false, b[t],
                                                       (short)0, acc[t], false, false);
  }

  #pragma unroll
  for (int t = 0; t < 8; ++t) {
    const int n = t * 16 + r;
    const float bs = Ub[n];
    #pragma unroll
    for (int v = 0; v < 8; ++v) {
      const int node = n0 + v + kb;
      if (node < N_NODES) {
        const size_t off = (size_t)node * EMB + n;
        const float nh = h[off] + lrelu(acc[t][v] + bs);
        h[off] = nh;
        hbf[off] = (__bf16)nh;
      }
    }
  }
}

__global__ void pool_kernel(const float* __restrict__ h, const int* __restrict__ batch,
                            float* __restrict__ sums, float* __restrict__ cnts) {
  const long idx = (long)blockIdx.x * blockDim.x + threadIdx.x;
  if (idx >= (long)N_NODES * EMB) return;
  const int node = (int)(idx >> 7);
  const int c = (int)(idx & 127);
  const int g = batch[node];
  atomicAdd(&sums[(size_t)g * EMB + c], h[idx]);
  if (c == 0) atomicAdd(&cnts[g], 1.0f);
}

__global__ void final_kernel(const float* __restrict__ sums, const float* __restrict__ cnts,
                             const float* __restrict__ pw, const float* __restrict__ pb,
                             float* __restrict__ out) {
  const int g = threadIdx.x;
  if (g >= NUM_GRAPHS) return;
  float cnt = cnts[g]; cnt = cnt > 1.0f ? cnt : 1.0f;
  float s = pb[0];
  #pragma unroll 4
  for (int c = 0; c < EMB; ++c) s += (sums[(size_t)g * EMB + c] / cnt) * pw[c];
  out[g] = s;
}

extern "C" void kernel_launch(void* const* d_in, const int* in_sizes, int n_in,
                              void* d_out, int out_size, void* d_ws, size_t ws_size,
                              hipStream_t stream) {
  (void)in_sizes; (void)n_in; (void)out_size; (void)ws_size;
  const float* x     = (const float*)d_in[0];
  const int*   eidx  = (const int*)d_in[1];
  const float* eattr = (const float*)d_in[2];
  const int*   batch = (const int*)d_in[3];
  const float* lin_w = (const float*)d_in[4];
  const float* lin_b = (const float*)d_in[5];
  const float* M1w   = (const float*)d_in[6];
  const float* M1b   = (const float*)d_in[7];
  const float* M2w   = (const float*)d_in[8];
  const float* M2b   = (const float*)d_in[9];
  const float* Uw    = (const float*)d_in[10];
  const float* Ub    = (const float*)d_in[11];
  const float* pw    = (const float*)d_in[12];
  const float* pb    = (const float*)d_in[13];
  const int* srcI = eidx;             // edge_index[0]
  const int* dstI = eidx + N_EDGES;   // edge_index[1]

  char* wsp = (char*)d_ws;
  float*  h    = (float*)wsp;  wsp += (size_t)N_NODES * EMB * 4;
  float*  aggr = (float*)wsp;  wsp += (size_t)N_NODES * EMB * 4;
  __bf16* hbf  = (__bf16*)wsp; wsp += (size_t)N_NODES * EMB * 2;
  __bf16* M1p  = (__bf16*)wsp; wsp += (size_t)N_LAYERS * EMB * 3 * EMB * 2;
  __bf16* M2p  = (__bf16*)wsp; wsp += (size_t)N_LAYERS * EMB * EMB * 2;
  __bf16* Up   = (__bf16*)wsp; wsp += (size_t)N_LAYERS * EMB * 2 * EMB * 2;
  float*  sums = (float*)wsp;  wsp += (size_t)NUM_GRAPHS * EMB * 4;
  float*  cnts = (float*)wsp;  wsp += (size_t)NUM_GRAPHS * 4;

  pack_kernel<<<192, 256, 0, stream>>>(M1w, M1p, N_LAYERS, EMB, 3 * EMB);
  pack_kernel<<<64,  256, 0, stream>>>(M2w, M2p, N_LAYERS, EMB, EMB);
  pack_kernel<<<128, 256, 0, stream>>>(Uw,  Up,  N_LAYERS, EMB, 2 * EMB);

  lin_in_kernel<<<N_NODES, EMB, 0, stream>>>(x, lin_w, lin_b, h, hbf);

  const int edgeBlocks = (N_EDGES + 127) / 128;
  const int nodeBlocks = (N_NODES + 127) / 128;
  for (int l = 0; l < N_LAYERS; ++l) {
    zero_kernel<<<1024, 256, 0, stream>>>(aggr, (long)N_NODES * EMB);
    edge_msg_kernel<<<edgeBlocks, 256, 0, stream>>>(
        hbf, eattr, srcI, dstI,
        M1p + (size_t)l * EMB * 3 * EMB, M1b + (size_t)l * EMB,
        M2p + (size_t)l * EMB * EMB,     M2b + (size_t)l * EMB,
        aggr);
    node_update_kernel<<<nodeBlocks, 256, 0, stream>>>(
        h, hbf, aggr, Up + (size_t)l * EMB * 2 * EMB, Ub + (size_t)l * EMB);
  }

  zero_kernel<<<16, 256, 0, stream>>>(sums, (long)NUM_GRAPHS * EMB + NUM_GRAPHS);
  pool_kernel<<<((long)N_NODES * EMB + 255) / 256, 256, 0, stream>>>(h, batch, sums, cnts);
  final_kernel<<<1, 64, 0, stream>>>(sums, cnts, pw, pb, (float*)d_out);
}
